// BlockedF8Linear_927712936105
// MI455X (gfx1250) — matrix operand order repaired and checked by reading, not compiled, not run on hardware
//
#include <hip/hip_runtime.h>

#if __has_include(<hip/amd_detail/amd_gfx1250_TDM.h>)
#define TDM_6ARG 1
#else
#define TDM_6ARG 0
#endif

typedef _Float16 v16h __attribute__((ext_vector_type(16)));
typedef _Float16 v8h  __attribute__((ext_vector_type(8)));
typedef _Float16 v4h  __attribute__((ext_vector_type(4)));
typedef float    v8f  __attribute__((ext_vector_type(8)));
typedef float    v4f  __attribute__((ext_vector_type(4)));
typedef unsigned int u32x4 __attribute__((ext_vector_type(4)));
typedef int      i32x4 __attribute__((ext_vector_type(4)));
typedef int      i32x8 __attribute__((ext_vector_type(8)));

#define KDIM   4096
#define NDIM   4096
#define SBLK   128
#define KC     64          // K-chunk per stage (2 WMMA k-steps)
#define LDSP   72          // padded LDS row stride in halves (144B = 36 banks)
#define TM     256         // M tile per workgroup
#define TN     128         // N tile per workgroup
#define A_H    (TM * LDSP) // halves per A buffer (18432 h = 36864 B)
#define B_H    (TN * LDSP) // halves per B buffer ( 9216 h = 18432 B)
#define SMEM_BYTES (2 * (A_H + B_H) * 2)   // 110592 B double-buffered

// ---------------- TDM (Tensor Data Mover) helpers ----------------
__device__ __forceinline__ unsigned lds_addr_of(const void* p) {
    // generic pointer to LDS: low 32 bits are the LDS byte offset
    return (unsigned)(unsigned long long)(uintptr_t)p;
}

// 2D fp16 tile load: tileRows x KC halves, row-major tensor with rowStride
// elements; LDS rows padded by 16B every 128B (72-half stride).
__device__ __forceinline__ void tdm_load_tile_f16(const _Float16* gsrc,
                                                  unsigned ldsOff, int tileRows,
                                                  int tensorCols, int rowStride) {
    unsigned long long ga = (unsigned long long)(uintptr_t)gsrc;
    u32x4 g0;
    g0.x = 1u;                                            // count=1, user desc
    g0.y = ldsOff;                                        // lds_addr
    g0.z = (unsigned)(ga & 0xffffffffull);                // global_addr[31:0]
    g0.w = (unsigned)((ga >> 32) & 0x01ffffffull)         // global_addr[56:32]
         | (2u << 30);                                    // type=2 ("image")
    i32x8 g1;
    g1[0] = (1 << 16)                                     // data_size = 2B
          | (1 << 20)                                     // pad_enable
          | (4 << 22)                                     // pad_interval: 32 DW (=128B row)
          | (3 << 25);                                    // pad_amount: 4 DW (=16B)
    g1[1] = (tensorCols & 0xffff) << 16;                  // tensor_dim0[15:0]
    g1[2] = ((tensorCols >> 16) & 0xffff)                 // tensor_dim0[31:16]
          | ((tileRows & 0xffff) << 16);                  // tensor_dim1[15:0]
    g1[3] = (KC << 16);                                   // tile_dim0 = 64 elems
    g1[4] = tileRows;                                     // tile_dim1
    g1[5] = rowStride;                                    // tensor_dim0_stride[31:0]
    g1[6] = 0;
    g1[7] = 0;
    i32x4 gz  = {0, 0, 0, 0};
#if TDM_6ARG
    i32x8 gz8 = {0, 0, 0, 0, 0, 0, 0, 0};
    __builtin_amdgcn_tensor_load_to_lds(g0, g1, gz, gz, gz8, 0);
#else
    __builtin_amdgcn_tensor_load_to_lds(g0, g1, gz, gz, 0);
#endif
}

// ---------------- prepass: fp32 -> fp16 (scale folded into W) ----------------
__global__ __launch_bounds__(256)
void cvt_x_kernel(const float* __restrict__ x, _Float16* __restrict__ xh, int n4) {
    int i = blockIdx.x * 256 + threadIdx.x;
    if (i >= n4) return;
    v4f v = ((const v4f*)x)[i];
    v4h h;
    h.x = (_Float16)v.x; h.y = (_Float16)v.y;
    h.z = (_Float16)v.z; h.w = (_Float16)v.w;
    ((v4h*)xh)[i] = h;
}

__global__ __launch_bounds__(256)
void cvt_w_kernel(const float* __restrict__ w, const float* __restrict__ sinv,
                  _Float16* __restrict__ wh, int n4) {
    int i = blockIdx.x * 256 + threadIdx.x;
    if (i >= n4) return;
    const int o    = i >> 10;             // (i*4) / 4096
    const int icol = (i & 1023) * 4;
    const float s  = sinv[(o >> 7) * (KDIM / SBLK) + (icol >> 7)];
    v4f v = ((const v4f*)w)[i];
    v4h h;
    h.x = (_Float16)(v.x * s); h.y = (_Float16)(v.y * s);
    h.z = (_Float16)(v.z * s); h.w = (_Float16)(v.w * s);
    ((v4h*)wh)[i] = h;
}

// ---------------- main GEMM: TDM double-buffered fp16 WMMA ----------------
__global__ __launch_bounds__(512)
void gemm_f16_tdm_kernel(const _Float16* __restrict__ xh,
                         const _Float16* __restrict__ wh,
                         const float* __restrict__ bias,
                         float* __restrict__ out, int M)
{
    extern __shared__ __align__(16) char smem[];
    _Float16* Abuf = (_Float16*)smem;                       // 2 x A_H halves
    _Float16* Bbuf = (_Float16*)(smem + 2 * A_H * 2);       // 2 x B_H halves

    const int tid   = threadIdx.x;
    const int lane  = tid & 31;
    const int wave  = tid >> 5;       // 16 waves
    const int waveM = wave & 7;       // 8 waves along M -> 32 rows each
    const int waveN = wave >> 3;      // 2 waves along N -> 64 cols each
    const int lrow  = lane & 15;
    const int hsel  = lane >> 4;

    const int n0 = blockIdx.x * TN;
    const int m0 = blockIdx.y * TM;

    v8f acc[2][4];
    #pragma unroll
    for (int mi = 0; mi < 2; ++mi)
        #pragma unroll
        for (int ni = 0; ni < 4; ++ni)
            acc[mi][ni] = (v8f){0.f,0.f,0.f,0.f,0.f,0.f,0.f,0.f};

    const int nChunks = KDIM / KC;    // 64
    // prologue: DMA chunk 0 into buffer 0 (one wave drives the TDM)
    if (wave == 0) {
        tdm_load_tile_f16(xh + (size_t)m0 * KDIM, lds_addr_of(Abuf), TM, KDIM, KDIM);
        tdm_load_tile_f16(wh + (size_t)n0 * KDIM, lds_addr_of(Bbuf), TN, KDIM, KDIM);
    }

    for (int kc = 0; kc < nChunks; ++kc) {
        const int cur = kc & 1;
        const int nxt = cur ^ 1;
        if (wave == 0) {
            if (kc + 1 < nChunks) {
                const int k1 = (kc + 1) * KC;
                tdm_load_tile_f16(xh + (size_t)m0 * KDIM + k1,
                                  lds_addr_of(Abuf + nxt * A_H), TM, KDIM, KDIM);
                tdm_load_tile_f16(wh + (size_t)n0 * KDIM + k1,
                                  lds_addr_of(Bbuf + nxt * B_H), TN, KDIM, KDIM);
                __builtin_amdgcn_s_wait_tensorcnt(2);  // current chunk's pair done
            } else {
                __builtin_amdgcn_s_wait_tensorcnt(0);
            }
        }
        __syncthreads();              // LDS data visible to all 16 waves

        const _Float16* As = Abuf + cur * A_H;
        const _Float16* Bs = Bbuf + cur * B_H;

        #pragma unroll
        for (int ksl = 0; ksl < 2; ++ksl) {
            const int kb = ksl * 32;
            // A fragment (16x32 f16): lane holds row M=lrow,
            // K chunks at hsel*8 and hsel*8+16.
            v16h afrag[2];
            #pragma unroll
            for (int mi = 0; mi < 2; ++mi) {
                const int r = waveM * 32 + mi * 16 + lrow;
                v8h lo = *(const v8h*)(As + r * LDSP + kb + hsel * 8);
                v8h hi = *(const v8h*)(As + r * LDSP + kb + hsel * 8 + 16);
                afrag[mi] = __builtin_shufflevector(lo, hi,
                    0,1,2,3,4,5,6,7,8,9,10,11,12,13,14,15);
            }
            // B fragment (32x16 f16): lane holds column N=lrow (= weight row),
            // K chunks at hsel*8 and hsel*8+16.
            v16h bfrag[4];
            #pragma unroll
            for (int ni = 0; ni < 4; ++ni) {
                const int r = waveN * 64 + ni * 16 + lrow;
                v8h lo = *(const v8h*)(Bs + r * LDSP + kb + hsel * 8);
                v8h hi = *(const v8h*)(Bs + r * LDSP + kb + hsel * 8 + 16);
                bfrag[ni] = __builtin_shufflevector(lo, hi,
                    0,1,2,3,4,5,6,7,8,9,10,11,12,13,14,15);
            }
            #pragma unroll
            for (int mi = 0; mi < 2; ++mi)
                #pragma unroll
                for (int ni = 0; ni < 4; ++ni)
                    acc[mi][ni] = __builtin_amdgcn_wmma_f32_16x16x32_f16(
                        false, afrag[mi], false, bfrag[ni],
                        (short)0, acc[mi][ni], false, false);
        }
        __syncthreads();              // reads done before buffer is re-DMA'd
    }

    // epilogue: C layout -> lane holds column lrow; VGPR j -> row j + hsel*8
    const int colBase = n0 + waveN * 64;
    float bvals[4];
    #pragma unroll
    for (int ni = 0; ni < 4; ++ni)
        bvals[ni] = bias[colBase + ni * 16 + lrow];

    #pragma unroll
    for (int mi = 0; mi < 2; ++mi) {
        const int rowBase = m0 + waveM * 32 + mi * 16 + hsel * 8;
        #pragma unroll
        for (int ni = 0; ni < 4; ++ni) {
            const int col = colBase + ni * 16 + lrow;
            #pragma unroll
            for (int j = 0; j < 8; ++j)
                out[(size_t)(rowBase + j) * NDIM + col] = acc[mi][ni][j] + bvals[ni];
        }
    }
}

// ---------------- fallback: fused fp32->fp16 single-pass (round-0 kernel) ----
__global__ __launch_bounds__(256)
void gemm_fused_kernel(const float* __restrict__ x, const float* __restrict__ w,
                       const float* __restrict__ sinv, const float* __restrict__ bias,
                       float* __restrict__ out, int M)
{
    __shared__ _Float16 As[128][LDSP];
    __shared__ _Float16 Bs[128][LDSP];

    const int tid = threadIdx.x, lane = tid & 31, wave = tid >> 5;
    const int waveM = wave & 3, waveN = wave >> 2;
    const int lrow = lane & 15, hsel = lane >> 4;
    const int n0 = blockIdx.x * 128, m0 = blockIdx.y * 128;
    const int oB = blockIdx.x;

    v8f acc[2][4];
    #pragma unroll
    for (int mi = 0; mi < 2; ++mi)
        #pragma unroll
        for (int ni = 0; ni < 4; ++ni)
            acc[mi][ni] = (v8f){0.f,0.f,0.f,0.f,0.f,0.f,0.f,0.f};

    const int nChunks = KDIM / KC;
    for (int kc = 0; kc < nChunks; ++kc) {
        const int k0 = kc * KC;
        const float s = sinv[oB * (KDIM / SBLK) + (k0 >> 7)];
        __syncthreads();
        #pragma unroll
        for (int i = 0; i < 8; ++i) {
            const int idx = tid + i * 256;
            const int r = idx >> 4, c4 = idx & 15;
            v4f av = *(const v4f*)(x + (size_t)(m0 + r) * KDIM + k0 + c4 * 4);
            v4f bv = *(const v4f*)(w + (size_t)(n0 + r) * KDIM + k0 + c4 * 4);
            v4h ah, bh;
            ah.x = (_Float16)av.x; ah.y = (_Float16)av.y;
            ah.z = (_Float16)av.z; ah.w = (_Float16)av.w;
            bh.x = (_Float16)(bv.x * s); bh.y = (_Float16)(bv.y * s);
            bh.z = (_Float16)(bv.z * s); bh.w = (_Float16)(bv.w * s);
            *(v4h*)&As[r][c4 * 4] = ah;
            *(v4h*)&Bs[r][c4 * 4] = bh;
        }
        __syncthreads();
        #pragma unroll
        for (int ksl = 0; ksl < 2; ++ksl) {
            const int kb = ksl * 32;
            v16h afrag[2];
            #pragma unroll
            for (int mi = 0; mi < 2; ++mi) {
                const int r = waveM * 32 + mi * 16 + lrow;
                v8h lo = *(const v8h*)&As[r][kb + hsel * 8];
                v8h hi = *(const v8h*)&As[r][kb + hsel * 8 + 16];
                afrag[mi] = __builtin_shufflevector(lo, hi,
                    0,1,2,3,4,5,6,7,8,9,10,11,12,13,14,15);
            }
            v16h bfrag[4];
            #pragma unroll
            for (int ni = 0; ni < 4; ++ni) {
                const int r = waveN * 64 + ni * 16 + lrow;
                v8h lo = *(const v8h*)&Bs[r][kb + hsel * 8];
                v8h hi = *(const v8h*)&Bs[r][kb + hsel * 8 + 16];
                bfrag[ni] = __builtin_shufflevector(lo, hi,
                    0,1,2,3,4,5,6,7,8,9,10,11,12,13,14,15);
            }
            #pragma unroll
            for (int mi = 0; mi < 2; ++mi)
                #pragma unroll
                for (int ni = 0; ni < 4; ++ni)
                    acc[mi][ni] = __builtin_amdgcn_wmma_f32_16x16x32_f16(
                        false, afrag[mi], false, bfrag[ni],
                        (short)0, acc[mi][ni], false, false);
        }
    }
    const int colBase = n0 + waveN * 64;
    float bvals[4];
    #pragma unroll
    for (int ni = 0; ni < 4; ++ni)
        bvals[ni] = bias[colBase + ni * 16 + lrow];
    #pragma unroll
    for (int mi = 0; mi < 2; ++mi) {
        const int rowBase = m0 + waveM * 32 + mi * 16 + hsel * 8;
        #pragma unroll
        for (int ni = 0; ni < 4; ++ni) {
            const int col = colBase + ni * 16 + lrow;
            #pragma unroll
            for (int j = 0; j < 8; ++j)
                out[(size_t)(rowBase + j) * NDIM + col] = acc[mi][ni][j] + bvals[ni];
        }
    }
}

extern "C" void kernel_launch(void* const* d_in, const int* in_sizes, int n_in,
                              void* d_out, int out_size, void* d_ws, size_t ws_size,
                              hipStream_t stream) {
    const float* x    = (const float*)d_in[0];
    const float* w    = (const float*)d_in[1];
    const float* sinv = (const float*)d_in[2];
    const float* bias = (const float*)d_in[3];
    float* out = (float*)d_out;

    const int M = in_sizes[0] / KDIM;    // 4096
    const size_t xhBytes = (size_t)M * KDIM * sizeof(_Float16);
    const size_t whBytes = (size_t)NDIM * KDIM * sizeof(_Float16);

    if (ws_size >= xhBytes + whBytes && (M % TM) == 0) {
        _Float16* xh = (_Float16*)d_ws;
        _Float16* wh = (_Float16*)((char*)d_ws + xhBytes);
        const int n4x = M * KDIM / 4;
        const int n4w = NDIM * KDIM / 4;
        cvt_x_kernel<<<(n4x + 255) / 256, 256, 0, stream>>>(x, xh, n4x);
        cvt_w_kernel<<<(n4w + 255) / 256, 256, 0, stream>>>(w, sinv, wh, n4w);
        dim3 grid(NDIM / TN, M / TM);    // 32 x 16
        gemm_f16_tdm_kernel<<<grid, dim3(512), SMEM_BYTES, stream>>>(xh, wh, bias, out, M);
    } else {
        dim3 grid(NDIM / 128, M / 128);
        gemm_fused_kernel<<<grid, dim3(256), 0, stream>>>(x, w, sinv, bias, out, M);
    }
}
